// TransformerBlock_74715251081242
// MI455X (gfx1250) — compile-verified, matrix operand order checked by reading, not run on hardware
//
#include <hip/hip_runtime.h>
#include <math.h>

// ---------------------------------------------------------------------------
// Types for CDNA5 WMMA (wave32): V_WMMA_F32_16X16X32_BF16
// ---------------------------------------------------------------------------
typedef __attribute__((ext_vector_type(16))) __bf16 v16bf;
typedef __attribute__((ext_vector_type(8)))  float  v8f;

typedef unsigned short u16;
typedef unsigned int   u32;
typedef unsigned long long u64;

// Problem constants (match reference)
#define BB   4
#define NN   1024
#define CC   1024
#define HH   16
#define HD   64
#define TD   256
#define C3   3072      // 3*C
#define FF   4096      // MULT*C
#define BN   4096      // B*N

__device__ __forceinline__ u16 f2bf(float f) {
    u32 x = __float_as_uint(f);
    u32 r = (x + 0x7FFFu + ((x >> 16) & 1u)) >> 16;
    return (u16)r;
}

// Load one 16x32 bf16 fragment (A layout per CDNA5 ISA; Bt rows use the same
// mirrored layout). base points at element (0,0) of a row-major tile with row
// stride `ldr` (in elements). Lane L: row = L&15, k = (L>>4)*8 + {0..7, 16..23}.
__device__ __forceinline__ v16bf load_frag(const u16* base, int ldr, int lane) {
    const int row  = lane & 15;
    const int half = (lane >> 4) & 1;
    const u16* p = base + (size_t)row * ldr + half * 8;
    union { uint4 u[2]; v16bf v; } f;
    f.u[0] = *reinterpret_cast<const uint4*>(p);
    f.u[1] = *reinterpret_cast<const uint4*>(p + 16);
    return f.v;
}

__device__ __forceinline__ v8f wmma_bf16(v16bf a, v16bf b, v8f c) {
    return __builtin_amdgcn_wmma_f32_16x16x32_bf16(
        false, a, false, b, (short)0, c, false, false);
}

// Async DMA: copy 16 bytes global -> LDS without touching VGPR data lanes.
// ldsOff is the wave-relative LDS byte offset (low 32 bits of flat address:
// the LDS aperture tag lives entirely in addr[63:32]).  Tracked by ASYNCcnt.
__device__ __forceinline__ void async_b128(u32 ldsOff, const void* gptr) {
    u64 ga = (u64)(uintptr_t)gptr;
    asm volatile("global_load_async_to_lds_b128 %0, %1, off"
                 :: "v"(ldsOff), "v"(ga) : "memory");
}

// ---------------------------------------------------------------------------
// Generic bf16 GEMM:  out = act(A @ Bt^T + bias) [+ res]
//   A  : M x K bf16 row-major
//   Bt : N x K bf16 row-major (i.e. B transposed)
//   Block tile 64(M) x 128(N), 4 waves (128 thr), wave tile 32x64,
//   K-chunks of 32 staged via double-buffered async global->LDS DMA.
//   blockIdx.z batching via element strides sA/sB/sO.
// ---------------------------------------------------------------------------
__global__ __launch_bounds__(128) void gemm_bf16_kernel(
    const u16* __restrict__ A, const u16* __restrict__ Bt,
    const float* __restrict__ bias, const float* __restrict__ res,
    float* __restrict__ outf, u16* __restrict__ outbf,
    int M, int N, int K, int act,
    long long sA, long long sB, long long sO)
{
    A  += (size_t)blockIdx.z * (size_t)sA;
    Bt += (size_t)blockIdx.z * (size_t)sB;
    const size_t outOff = (size_t)blockIdx.z * (size_t)sO;

    __shared__ __align__(16) u16 As[2][64 * 32];    // 2 x 4 KB
    __shared__ __align__(16) u16 Bs[2][128 * 32];   // 2 x 8 KB

    const int tid  = threadIdx.x;
    const int lane = tid & 31;
    const int wave = tid >> 5;
    const int wm   = (wave & 1) * 32;    // M offset of wave tile
    const int wn   = (wave >> 1) * 64;   // N offset of wave tile
    const int m0   = blockIdx.y * 64;
    const int n0   = blockIdx.x * 128;

    // per-thread staging coordinates: 4 b128 per 32-row tile section
    const int r4 = tid >> 2;             // 0..31
    const int c8 = (tid & 3) * 8;        // 0/8/16/24
    const u32 asBase = (u32)(uintptr_t)(const void*)&As[0][0];
    const u32 bsBase = (u32)(uintptr_t)(const void*)&Bs[0][0];

    const int steps = K >> 5;

    auto prefetch = [&](int slot, int k0) {
#pragma unroll
        for (int it = 0; it < 2; ++it) {          // A tile: 64 rows
            const int row = r4 + it * 32;
            async_b128(asBase + (u32)(slot * 4096 + (row * 32 + c8) * 2),
                       &A[(size_t)(m0 + row) * K + k0 + c8]);
        }
#pragma unroll
        for (int it = 0; it < 4; ++it) {          // B tile: 128 rows
            const int row = r4 + it * 32;
            async_b128(bsBase + (u32)(slot * 8192 + (row * 32 + c8) * 2),
                       &Bt[(size_t)(n0 + row) * K + k0 + c8]);
        }
    };

    v8f zf;
#pragma unroll
    for (int r = 0; r < 8; ++r) zf[r] = 0.0f;
    v8f acc[2][4];
#pragma unroll
    for (int i = 0; i < 2; ++i)
#pragma unroll
        for (int j = 0; j < 4; ++j) acc[i][j] = zf;

    prefetch(0, 0);
    for (int kc = 0; kc < steps; ++kc) {
        // issue next chunk (wraps to 0 on the final iteration: dummy load that
        // keeps the ASYNCcnt wait immediate uniform; never read)
        const int knext = (kc + 1 < steps) ? (kc + 1) * 32 : 0;
        prefetch((kc + 1) & 1, knext);

        // 12 async ops outstanding; retire the 6 belonging to this buffer.
        asm volatile("s_wait_asynccnt 0x6" ::: "memory");
        __syncthreads();

        const int slot = kc & 1;
        v16bf af0 = load_frag(&As[slot][(wm +  0) * 32], 32, lane);
        v16bf af1 = load_frag(&As[slot][(wm + 16) * 32], 32, lane);
#pragma unroll
        for (int j = 0; j < 4; ++j) {
            v16bf bfj = load_frag(&Bs[slot][(wn + j * 16) * 32], 32, lane);
            acc[0][j] = wmma_bf16(af0, bfj, acc[0][j]);
            acc[1][j] = wmma_bf16(af1, bfj, acc[1][j]);
        }
        __syncthreads();   // reads done before next iteration's DMA overwrites
    }

    const int colBase = lane & 15;
    const int rgrp    = lane >> 4;
#pragma unroll
    for (int i = 0; i < 2; ++i) {
#pragma unroll
        for (int j = 0; j < 4; ++j) {
            const int col = n0 + wn + j * 16 + colBase;
            const float bv = bias ? bias[col] : 0.0f;
#pragma unroll
            for (int r = 0; r < 8; ++r) {
                const int row = m0 + wm + i * 16 + r + 8 * rgrp;
                float v = acc[i][j][r] + bv;
                if (act == 1)  // exact GELU
                    v = 0.5f * v * (1.0f + erff(v * 0.70710678118654752f));
                if (res) v += res[(size_t)row * N + col];
                const size_t off = outOff + (size_t)row * N + col;
                if (outf)  outf[off]  = v;
                if (outbf) outbf[off] = f2bf(v);
            }
        }
    }
}

// ---------------------------------------------------------------------------
// LayerNorm over C=1024, output bf16 (GEMM A operand). 1 row per 256-thr block.
// ---------------------------------------------------------------------------
__global__ __launch_bounds__(256) void layernorm_kernel(
    const float* __restrict__ x, const float* __restrict__ g,
    const float* __restrict__ bvec, u16* __restrict__ out)
{
    const int row = blockIdx.x;
    const int tid = threadIdx.x;
    const float* xr = x + (size_t)row * CC;

    float v[4];
    float s = 0.0f;
#pragma unroll
    for (int i = 0; i < 4; ++i) { v[i] = xr[tid + i * 256]; s += v[i]; }

    __shared__ float red[256];
    red[tid] = s; __syncthreads();
    for (int st = 128; st > 0; st >>= 1) {
        if (tid < st) red[tid] += red[tid + st];
        __syncthreads();
    }
    const float mean = red[0] * (1.0f / (float)CC);
    __syncthreads();

    s = 0.0f;
#pragma unroll
    for (int i = 0; i < 4; ++i) { float d = v[i] - mean; s += d * d; }
    red[tid] = s; __syncthreads();
    for (int st = 128; st > 0; st >>= 1) {
        if (tid < st) red[tid] += red[tid + st];
        __syncthreads();
    }
    const float var  = red[0] * (1.0f / (float)CC);
    const float rstd = rsqrtf(var + 1e-5f);

#pragma unroll
    for (int i = 0; i < 4; ++i) {
        const int c = tid + i * 256;
        out[(size_t)row * CC + c] = f2bf((v[i] - mean) * rstd * g[c] + bvec[c]);
    }
}

// ---------------------------------------------------------------------------
// tqkv = t @ time_w + time_b   (B x 3C, tiny: K=256)
// ---------------------------------------------------------------------------
__global__ __launch_bounds__(256) void tqkv_kernel(
    const float* __restrict__ t, const float* __restrict__ tw,
    const float* __restrict__ tb, float* __restrict__ tq)
{
    const int tid = blockIdx.x * 256 + threadIdx.x;     // 0 .. B*3C-1
    const int b = tid / C3;
    const int c = tid % C3;
    float s = tb[c];
    for (int k = 0; k < TD; ++k)
        s += t[b * TD + k] * tw[(size_t)k * C3 + c];
    tq[tid] = s;
}

// ---------------------------------------------------------------------------
// Split qkv (fp32, bias already added) + per-batch time bias into
//   q_bf [b,h,n,d], k_bf [b,h,n,d], vt_bf [b,h,d,n]   (bf16)
// ---------------------------------------------------------------------------
__global__ __launch_bounds__(256) void build_qkv_kernel(
    const float* __restrict__ qkvf, const float* __restrict__ tq,
    u16* __restrict__ q_bf, u16* __restrict__ k_bf, u16* __restrict__ vt_bf)
{
    const size_t tid = (size_t)blockIdx.x * 256 + threadIdx.x;  // B*H*N*HD
    const int d = (int)(tid & 63);
    const int n = (int)((tid >> 6) & 1023);
    const int h = (int)((tid >> 16) & 15);
    const int b = (int)(tid >> 20);
    const size_t row = (size_t)b * NN + n;
    const int col = h * HD + d;

    const float qf = qkvf[row * C3 +        col] + tq[b * C3 +        col];
    const float kf = qkvf[row * C3 + CC   + col] + tq[b * C3 + CC   + col];
    const float vf = qkvf[row * C3 + 2*CC + col] + tq[b * C3 + 2*CC + col];

    const size_t bh = (size_t)(b * HH + h);
    q_bf [(bh * NN + n) * HD + d] = f2bf(qf);
    k_bf [(bh * NN + n) * HD + d] = f2bf(kf);
    vt_bf[(bh * HD + d) * NN + n] = f2bf(vf);
}

// ---------------------------------------------------------------------------
// Attention for one batch b. Grid (N/16, H), 1 wave (32 thr) per block.
// Streaming (online-softmax) over m in 32-wide chunks; rel bias read from
// R[h] = q @ Er^T with the exact _skew indexing:
//   m <= n  -> R[n,   m-n+1023]
//   m == n+1-> 0
//   m >  n+1-> R[n+1, m-n-2]
// ---------------------------------------------------------------------------
__global__ __launch_bounds__(32) void attention_kernel(
    const u16* __restrict__ q_bf, const u16* __restrict__ k_bf,
    const u16* __restrict__ vt_bf, const float* __restrict__ Rbuf,
    u16* __restrict__ attnA, int b)
{
    const int lane = threadIdx.x & 31;
    const int h    = blockIdx.y;
    const int n0   = blockIdx.x * 16;

    const u16* qb  = q_bf  + ((size_t)(b * HH + h) * NN) * HD;
    const u16* kb  = k_bf  + ((size_t)(b * HH + h) * NN) * HD;
    const u16* vtb = vt_bf + ((size_t)(b * HH + h) * HD) * NN;
    const float* Rh = Rbuf + (size_t)h * NN * NN;

    // q tile fragments (16 rows x K=64 -> two K-chunks), kept for whole loop
    const v16bf aq0 = load_frag(qb + (size_t)n0 * HD +  0, HD, lane);
    const v16bf aq1 = load_frag(qb + (size_t)n0 * HD + 32, HD, lane);

    v8f zf;
#pragma unroll
    for (int r = 0; r < 8; ++r) zf[r] = 0.0f;
    v8f o[4] = { zf, zf, zf, zf };
    float mrow[8], lrow[8];
#pragma unroll
    for (int r = 0; r < 8; ++r) { mrow[r] = -3.0e38f; lrow[r] = 0.0f; }

    __shared__ __align__(16) u16 Pt[16 * 32];

    const int colA = lane & 15;
    const int rgrp = lane >> 4;

    for (int m0 = 0; m0 < NN; m0 += 32) {
        // S tiles: cols m0..m0+15 (s0) and m0+16..m0+31 (s1), K = HD = 64
        v8f s0 = zf, s1 = zf;
        {
            v16bf bk;
            bk = load_frag(kb + (size_t)(m0     ) * HD +  0, HD, lane); s0 = wmma_bf16(aq0, bk, s0);
            bk = load_frag(kb + (size_t)(m0     ) * HD + 32, HD, lane); s0 = wmma_bf16(aq1, bk, s0);
            bk = load_frag(kb + (size_t)(m0 + 16) * HD +  0, HD, lane); s1 = wmma_bf16(aq0, bk, s1);
            bk = load_frag(kb + (size_t)(m0 + 16) * HD + 32, HD, lane); s1 = wmma_bf16(aq1, bk, s1);
        }

        float alpha[8];
#pragma unroll
        for (int r = 0; r < 8; ++r) {
            const int n = n0 + r + 8 * rgrp;

            int m = m0 + colA;
            float rel0, rel1;
            if (m <= n)          rel0 = Rh[(size_t)n * NN + (m - n + NN - 1)];
            else if (m == n + 1) rel0 = 0.0f;
            else                 rel0 = Rh[(size_t)(n + 1) * NN + (m - n - 2)];
            m = m0 + 16 + colA;
            if (m <= n)          rel1 = Rh[(size_t)n * NN + (m - n + NN - 1)];
            else if (m == n + 1) rel1 = 0.0f;
            else                 rel1 = Rh[(size_t)(n + 1) * NN + (m - n - 2)];

            const float v0 = s0[r] * 0.125f + rel0;   // HD^-0.5 = 1/8
            const float v1 = s1[r] * 0.125f + rel1;

            // row max across the 16-lane half-group
            float tmax = fmaxf(v0, v1);
            tmax = fmaxf(tmax, __shfl_xor(tmax, 1, 32));
            tmax = fmaxf(tmax, __shfl_xor(tmax, 2, 32));
            tmax = fmaxf(tmax, __shfl_xor(tmax, 4, 32));
            tmax = fmaxf(tmax, __shfl_xor(tmax, 8, 32));

            const float newm = fmaxf(mrow[r], tmax);
            const float a    = __expf(mrow[r] - newm);
            const float p0   = __expf(v0 - newm);
            const float p1   = __expf(v1 - newm);
            float rs = p0 + p1;
            rs += __shfl_xor(rs, 1, 32);
            rs += __shfl_xor(rs, 2, 32);
            rs += __shfl_xor(rs, 4, 32);
            rs += __shfl_xor(rs, 8, 32);

            lrow[r]  = lrow[r] * a + rs;
            mrow[r]  = newm;
            alpha[r] = a;

            const int prow = r + 8 * rgrp;
            Pt[prow * 32 +      colA] = f2bf(p0);
            Pt[prow * 32 + 16 + colA] = f2bf(p1);
        }

        // rescale accumulators
#pragma unroll
        for (int dj = 0; dj < 4; ++dj)
#pragma unroll
            for (int r = 0; r < 8; ++r) o[dj][r] *= alpha[r];

        __syncthreads();
        const v16bf ap = load_frag(Pt, 32, lane);        // P tile 16x32 as A
#pragma unroll
        for (int dj = 0; dj < 4; ++dj) {
            const v16bf bv = load_frag(vtb + (size_t)(dj * 16) * NN + m0, NN, lane);
            o[dj] = wmma_bf16(ap, bv, o[dj]);
        }
        __syncthreads();
    }

    // finalize: divide by softmax denom, write bf16 into concat-head layout
#pragma unroll
    for (int dj = 0; dj < 4; ++dj) {
#pragma unroll
        for (int r = 0; r < 8; ++r) {
            const int n   = n0 + r + 8 * rgrp;
            const int col = h * HD + dj * 16 + colA;
            attnA[((size_t)(b * NN + n)) * CC + col] = f2bf(o[dj][r] / lrow[r]);
        }
    }
}

// ---------------------------------------------------------------------------
// Weight conversion: Wt[n*K+k] = bf16(W[k*N+n])   (pre-transpose for GEMM B)
// ---------------------------------------------------------------------------
__global__ __launch_bounds__(256) void convT_kernel(
    const float* __restrict__ W, u16* __restrict__ Wt, int K, int N)
{
    const size_t tid = (size_t)blockIdx.x * 256 + threadIdx.x;
    if (tid >= (size_t)K * N) return;
    const size_t n = tid / K, k = tid % K;
    Wt[tid] = f2bf(W[k * (size_t)N + n]);
}

__global__ __launch_bounds__(256) void conv_kernel(
    const float* __restrict__ in, u16* __restrict__ out, int n)
{
    const int tid = blockIdx.x * 256 + threadIdx.x;
    if (tid < n) out[tid] = f2bf(in[tid]);
}

// ---------------------------------------------------------------------------
// Host-side orchestration
// ---------------------------------------------------------------------------
extern "C" void kernel_launch(void* const* d_in, const int* in_sizes, int n_in,
                              void* d_out, int out_size, void* d_ws, size_t ws_size,
                              hipStream_t stream)
{
    const float* x      = (const float*)d_in[0];
    const float* t      = (const float*)d_in[1];
    const float* ln1_g  = (const float*)d_in[2];
    const float* ln1_b  = (const float*)d_in[3];
    const float* qkv_w  = (const float*)d_in[4];
    const float* qkv_b  = (const float*)d_in[5];
    const float* time_w = (const float*)d_in[6];
    const float* time_b = (const float*)d_in[7];
    const float* out_w  = (const float*)d_in[8];
    const float* out_b  = (const float*)d_in[9];
    const float* Er     = (const float*)d_in[10];
    const float* ln2_g  = (const float*)d_in[11];
    const float* ln2_b  = (const float*)d_in[12];
    const float* ff1_w  = (const float*)d_in[13];
    const float* ff1_b  = (const float*)d_in[14];
    const float* ff2_w  = (const float*)d_in[15];
    const float* ff2_b  = (const float*)d_in[16];
    float* outp = (float*)d_out;

    char* ws = (char*)d_ws;
    size_t off = 0;
    auto carve = [&](size_t bytes) -> void* {
        void* p = ws + off;
        off = (off + bytes + 255) & ~(size_t)255;
        return p;
    };

    u16*   wt_qkv = (u16*)  carve((size_t)C3 * CC * 2);
    u16*   wt_out = (u16*)  carve((size_t)CC * CC * 2);
    u16*   wt_ff1 = (u16*)  carve((size_t)FF * CC * 2);
    u16*   wt_ff2 = (u16*)  carve((size_t)CC * FF * 2);
    u16*   er_bf  = (u16*)  carve((size_t)NN * HD * 2);
    u16*   xn_bf  = (u16*)  carve((size_t)BN * CC * 2);
    float* tq     = (float*)carve((size_t)BB * C3 * 4);
    float* qkvf   = (float*)carve((size_t)BN * C3 * 4);
    u16*   q_bf   = (u16*)  carve((size_t)BB * HH * NN * HD * 2);
    u16*   k_bf   = (u16*)  carve((size_t)BB * HH * NN * HD * 2);
    u16*   vt_bf  = (u16*)  carve((size_t)BB * HH * NN * HD * 2);
    float* Rbuf   = (float*)carve((size_t)HH * NN * NN * 4);   // reused per batch
    u16*   attnA  = (u16*)  carve((size_t)BN * CC * 2);
    float* x1     = (float*)carve((size_t)BN * CC * 4);
    u16*   ln2bf  = (u16*)  carve((size_t)BN * CC * 2);
    u16*   h_bf   = (u16*)  carve((size_t)BN * FF * 2);
    (void)in_sizes; (void)n_in; (void)out_size; (void)ws_size;

    // --- weight conversions (fp32 -> bf16, transposed to NxK) ---
    convT_kernel<<<(C3 * CC) / 256, 256, 0, stream>>>(qkv_w, wt_qkv, CC, C3);
    convT_kernel<<<(CC * CC) / 256, 256, 0, stream>>>(out_w, wt_out, CC, CC);
    convT_kernel<<<(FF * CC) / 256, 256, 0, stream>>>(ff1_w, wt_ff1, CC, FF);
    convT_kernel<<<(CC * FF) / 256, 256, 0, stream>>>(ff2_w, wt_ff2, FF, CC);
    conv_kernel <<<(NN * HD) / 256, 256, 0, stream>>>(Er, er_bf, NN * HD);

    // --- LN1, time-qkv ---
    layernorm_kernel<<<BN, 256, 0, stream>>>(x, ln1_g, ln1_b, xn_bf);
    tqkv_kernel<<<(BB * C3) / 256, 256, 0, stream>>>(t, time_w, time_b, tq);

    // --- QKV GEMM: (4096 x 3072 x 1024), +qkv_b, fp32 out ---
    gemm_bf16_kernel<<<dim3(C3 / 128, BN / 64, 1), 128, 0, stream>>>(
        xn_bf, wt_qkv, qkv_b, nullptr, qkvf, nullptr,
        BN, C3, CC, 0, 0, 0, 0);

    // --- split into q/k/v^T bf16 with time bias ---
    build_qkv_kernel<<<(BB * HH * NN * HD) / 256, 256, 0, stream>>>(
        qkvf, tq, q_bf, k_bf, vt_bf);

    // --- attention per batch (R buffer reused; stream serializes) ---
    for (int b = 0; b < BB; ++b) {
        // R[h] = q[b,h] @ Er^T   (M=1024, N=1024, K=64, batched over h)
        gemm_bf16_kernel<<<dim3(NN / 128, NN / 64, HH), 128, 0, stream>>>(
            q_bf + (size_t)b * HH * NN * HD, er_bf, nullptr, nullptr,
            Rbuf, nullptr, NN, NN, HD, 0,
            (long long)NN * HD, 0, (long long)NN * NN);

        attention_kernel<<<dim3(NN / 16, HH), 32, 0, stream>>>(
            q_bf, k_bf, vt_bf, Rbuf, attnA, b);
    }

    // --- output projection + residual: x1 = x + attnA @ out_w + out_b ---
    gemm_bf16_kernel<<<dim3(CC / 128, BN / 64, 1), 128, 0, stream>>>(
        attnA, wt_out, out_b, x, x1, nullptr,
        BN, CC, CC, 0, 0, 0, 0);

    // --- LN2 ---
    layernorm_kernel<<<BN, 256, 0, stream>>>(x1, ln2_g, ln2_b, ln2bf);

    // --- FF1 + exact GELU -> bf16 ---
    gemm_bf16_kernel<<<dim3(FF / 128, BN / 64, 1), 128, 0, stream>>>(
        ln2bf, wt_ff1, ff1_b, nullptr, nullptr, h_bf,
        BN, FF, CC, 1, 0, 0, 0);

    // --- FF2 + residual -> d_out ---
    gemm_bf16_kernel<<<dim3(CC / 128, BN / 64, 1), 128, 0, stream>>>(
        h_bf, wt_ff2, ff2_b, x1, outp, nullptr,
        BN, CC, FF, 0, 0, 0, 0);
}
